// GuidedMoEBasic_4372276707575
// MI455X (gfx1250) — matrix operand order repaired and checked by reading, not compiled, not run on hardware
//
#include <hip/hip_runtime.h>

// ---------------- problem constants ----------------
#define BB      8
#define LL      128
#define HH      768
#define NEMOC   7
#define DD      776          // H + NEMO + 1
#define DPAD    800          // DD padded to 25 K-chunks of 32
#define KCH     50           // K chunks of 32 over the 1600-wide padded pair dim
#define NCT     34           // column tiles of 16: 512 hidden (2 experts x 256) + gate(2) + pad
#define HID     256
#define NPAIR_B 8256         // L*(L+1)/2
#define NPAIR   (BB * NPAIR_B)  // 66048 = 64 * 1032 exactly

typedef __attribute__((ext_vector_type(16))) __bf16       v16bf;
typedef __attribute__((ext_vector_type(8)))  float        v8f;
typedef __attribute__((ext_vector_type(4)))  unsigned int u32x4;

static __device__ __forceinline__ unsigned short f2bf(float f) {
    unsigned int u = __builtin_bit_cast(unsigned int, f);
    unsigned int r = (u + 0x7FFFu + ((u >> 16) & 1u)) >> 16;   // RNE
    return (unsigned short)r;
}

// ---- kernel 1: fused concat -> bf16 cat[B,L,DPAD], zero-padded ----
__global__ void build_cat(const float* __restrict__ hp, const float* __restrict__ emo,
                          const float* __restrict__ spk, unsigned short* __restrict__ cat) {
    int idx = blockIdx.x * 256 + threadIdx.x;
    if (idx >= BB * LL * DPAD) return;
    int k  = idx % DPAD;
    int bl = idx / DPAD;
    float v = 0.f;
    if (k < HH)              v = hp[bl * HH + k];
    else if (k < HH + NEMOC) v = emo[bl * NEMOC + (k - HH)];
    else if (k == DD - 1)    v = spk[bl];
    cat[idx] = f2bf(v);
}

// ---- kernel 2: pre-swizzle stacked B = [W1_e0 | W1_e1 | W_gate | 0] into
//      per-lane WMMA B-fragments: bpk[c][ct][lane][16]  (bf16)
//      dense 16-bit B layout: lanes 0-15 hold K {0..7,16..23}; lanes 16-31 hold K {8..15,24..31}
__global__ void pack_w(const float* __restrict__ W1, const float* __restrict__ Wg,
                       unsigned short* __restrict__ bpk) {
    int idx = blockIdx.x * 256 + threadIdx.x;
    const int TOT = KCH * NCT * 512;
    if (idx >= TOT) return;
    int c    = idx / (NCT * 512);
    int r    = idx % (NCT * 512);
    int ct   = r / 512;
    int r2   = r % 512;
    int lane = r2 >> 4;
    int i    = r2 & 15;
    int klocal = ((lane < 16) ? 0 : 8) + ((i < 8) ? i : (8 + i));
    int kg     = c * 32 + klocal;                 // 0..1599 padded pair-dim row
    int ncol   = ct * 16 + (lane & 15);           // 0..543
    int part   = (kg >= DPAD) ? 1 : 0;
    int ksrc   = part ? (kg - DPAD) : kg;
    float v = 0.f;
    if (ksrc < DD) {
        int dprow = part ? (DD + ksrc) : ksrc;    // row in the original 1552 pair dim
        if (ncol < 512) {                          // expert hidden columns
            int e = ncol >> 8, h = ncol & 255;
            v = W1[((size_t)e * (2 * DD) + dprow) * HID + h];
        } else if (ncol < 514) {                   // gate columns
            v = Wg[dprow * 2 + (ncol - 512)];
        }
    }
    bpk[idx] = f2bf(v);
}

// ---- kernel 3: main WMMA GEMM + fused epilogue ----
__global__ __launch_bounds__(256) void moe_wmma(
    const unsigned short* __restrict__ cat,
    const unsigned short* __restrict__ bpk,
    const float* __restrict__ bg,
    const float* __restrict__ b1,
    const float* __restrict__ W2,
    const float* __restrict__ b2,
    float* __restrict__ out)
{
    __shared__ float hidL[16 * 533];
    __shared__ float partial[256];
    __shared__ float vals[64];

    const int tid  = threadIdx.x;
    const int lane = tid & 31;
    const int wave = tid >> 5;
    const int mt   = wave >> 1;    // M-tile 0..3 (16 pairs each)
    const int cs   = wave & 1;     // column stagger: tiles cs, cs+2, ...

    // keep the packed weights warm (global_prefetch_b8)
    __builtin_prefetch(bpk + (size_t)tid * 64, 0, 1);

    // decode this lane's pair index -> (batch, t, end_t)
    int n = blockIdx.x * 64 + mt * 16 + (lane & 15);
    int b = n / NPAIR_B;
    int p = n - b * NPAIR_B;
    int e = (int)((sqrtf(8.f * (float)p + 1.f) - 1.f) * 0.5f);
    while ((e + 1) * (e + 2) / 2 <= p) e++;
    while (e * (e + 1) / 2 > p)       e--;
    int t = p - e * (e + 1) / 2;
    const unsigned short* rowT = cat + ((size_t)(b * LL + t) * DPAD);
    const unsigned short* rowE = cat + ((size_t)(b * LL + e) * DPAD);

    // 16-bit A layout: lanes 0-15 hold K {0..7,16..23}; lanes 16-31 hold K {8..15,24..31}
    const int aoff = (lane < 16) ? 0 : 8;

    v8f acc[17];
    v8f z = {0.f, 0.f, 0.f, 0.f, 0.f, 0.f, 0.f, 0.f};
    #pragma unroll
    for (int i = 0; i < 17; ++i) acc[i] = z;

    for (int c = 0; c < KCH; ++c) {
        // a 32-wide K chunk never straddles the two utterances (25 chunks each)
        const unsigned short* src = (c < 25) ? rowT : rowE;
        int k0 = ((c < 25) ? c : (c - 25)) * 32;
        union { u32x4 u[2]; v16bf v; } A;
        A.u[0] = *(const u32x4*)(src + k0 + aoff);
        A.u[1] = *(const u32x4*)(src + k0 + aoff + 16);
        #pragma unroll
        for (int ti = 0; ti < 17; ++ti) {
            int ct = cs + 2 * ti;                       // <= 33; tile 33 is zero pad
            const unsigned short* bp = bpk + ((size_t)((c * NCT + ct) * 32 + lane) << 4);
            union { u32x4 u[2]; v16bf v; } Bf;
            Bf.u[0] = *(const u32x4*)(bp);
            Bf.u[1] = *(const u32x4*)(bp + 8);
            acc[ti] = __builtin_amdgcn_wmma_f32_16x16x32_bf16(
                false, A.v, false, Bf.v, (short)0, acc[ti], false, false);
        }
    }

    // fused epilogue, one 16-row M-tile group at a time (34 KB LDS reuse)
    for (int mg = 0; mg < 4; ++mg) {
        if (mt == mg) {
            int rbase = (lane < 16) ? 0 : 8;            // C layout: V j -> M=j / M=j+8
            #pragma unroll
            for (int ti = 0; ti < 17; ++ti) {
                int ct = cs + 2 * ti;
                if (ct < 33) {
                    int col = ct * 16 + (lane & 15);
                    #pragma unroll
                    for (int j = 0; j < 8; ++j)
                        hidL[(rbase + j) * 533 + col] = acc[ti][j];
                }
            }
        }
        __syncthreads();
        {   // 64 dot-products (16 rows x 2 experts x 2 outs), 4 threads each
            int task = tid & 63, q = tid >> 6;
            int row = task >> 2, ee = (task >> 1) & 1, oo = task & 1;
            float s = 0.f;
            int hb = ee * 256 + q * 64;
            const float* hrow = &hidL[row * 533];
            #pragma unroll 4
            for (int h = 0; h < 64; ++h) {
                int hc = hb + h;                         // == flat (e,h) index
                s += (hrow[hc] + b1[hc]) * W2[hc * 2 + oo];
            }
            partial[tid] = s;
        }
        __syncthreads();
        if (tid < 64) {
            int task = tid;
            int row = task >> 2, ee = (task >> 1) & 1, oo = task & 1;
            float tot = partial[task] + partial[task + 64] + partial[task + 128]
                      + partial[task + 192] + b2[ee * 2 + oo];
            float gate = hidL[row * 533 + 512 + ee] + bg[ee];
            vals[task] = gate * tot;
        }
        __syncthreads();
        if (tid < 64 && !(tid & 2)) {                    // sum the two experts
            int row = tid >> 2, oo = tid & 1;
            int nn = blockIdx.x * 64 + mg * 16 + row;
            out[nn * 2 + oo] = vals[tid] + vals[tid | 2];
        }
        __syncthreads();
    }
}

extern "C" void kernel_launch(void* const* d_in, const int* in_sizes, int n_in,
                              void* d_out, int out_size, void* d_ws, size_t ws_size,
                              hipStream_t stream) {
    const float* emo = (const float*)d_in[0];
    const float* hp  = (const float*)d_in[1];
    /* d_in[2] = input_ids (int64) unused by reference math */
    const float* spk = (const float*)d_in[3];
    const float* Wg  = (const float*)d_in[4];
    const float* bg  = (const float*)d_in[5];
    const float* W1  = (const float*)d_in[6];
    const float* b1  = (const float*)d_in[7];
    const float* W2  = (const float*)d_in[8];
    const float* b2  = (const float*)d_in[9];
    float* out = (float*)d_out;

    unsigned short* cat = (unsigned short*)d_ws;                 // 819,200 * 2 B
    unsigned short* bpk = cat + (size_t)BB * LL * DPAD;          // 870,400 * 2 B

    build_cat<<<(BB * LL * DPAD + 255) / 256, 256, 0, stream>>>(hp, emo, spk, cat);
    pack_w<<<(KCH * NCT * 512 + 255) / 256, 256, 0, stream>>>(W1, Wg, bpk);
    moe_wmma<<<NPAIR / 64, 256, 0, stream>>>(cat, bpk, bg, b1, W2, b2, out);
}
